// PretrainedEmbedding_87316685128290
// MI455X (gfx1250) — matrix operand order repaired and checked by reading, not compiled, not run on hardware
//
#include <hip/hip_runtime.h>
#include <stdint.h>

typedef __bf16 bf16;
typedef __attribute__((ext_vector_type(8)))  __bf16 v8bf;
typedef __attribute__((ext_vector_type(16))) __bf16 v16bf;
typedef __attribute__((ext_vector_type(8)))  float  v8f;
typedef __attribute__((ext_vector_type(4)))  unsigned int u32x4;
typedef __attribute__((ext_vector_type(8)))  int i32x8;
typedef __attribute__((ext_vector_type(4)))  int i32x4;
typedef uint32_t u32;
typedef uint64_t u64;

#define VOCAB  30522
#define HIDDEN 768
#define EMBED  512
#define BB     16
#define SS     512
#define WW     4
#define MM     (BB*SS)   // 8192

// ---------------- Kernel 1: per-batch stable-compaction scan ----------------
__global__ __launch_bounds__(512) void scan_valid(const int* __restrict__ words,
                                                  int* __restrict__ dest,
                                                  int* __restrict__ lens) {
  __shared__ int sb[SS];
  const int b = blockIdx.x;
  const int s = threadIdx.x;
  const int4 w = reinterpret_cast<const int4*>(words)[b * SS + s];
  const int valid = ((w.x | w.y | w.z | w.w) != 0) ? 1 : 0;  // ids are >= 0
  sb[s] = valid;
  __syncthreads();
  for (int off = 1; off < SS; off <<= 1) {
    int v = sb[s];
    if (s >= off) v += sb[s - off];
    __syncthreads();
    sb[s] = v;
    __syncthreads();
  }
  const int incl = sb[s];
  dest[b * SS + s] = valid ? (incl - valid) : -1;
  if (s == SS - 1) lens[b] = incl;
}

// ------- Kernel 2: gather + mean-pool + compacting scatter, bf16 hi/lo ------
__global__ __launch_bounds__(192) void pool_scatter(const int* __restrict__ words,
                                                    const float* __restrict__ embed,
                                                    const int* __restrict__ dest,
                                                    const int* __restrict__ lens,
                                                    bf16* __restrict__ xh,
                                                    bf16* __restrict__ xl) {
  const int bs = blockIdx.x;
  const int b  = bs >> 9;
  const int s  = bs & (SS - 1);
  const int4 w = reinterpret_cast<const int4*>(words)[bs];
  const int h  = threadIdx.x * 4;       // 192*4 = 768

  float v0 = 0.f, v1 = 0.f, v2 = 0.f, v3 = 0.f;
  int cnt = 0;
  auto add = [&](int id) {
    const float4 e = *reinterpret_cast<const float4*>(embed + (size_t)id * HIDDEN + h);
    v0 += e.x; v1 += e.y; v2 += e.z; v3 += e.w;
    ++cnt;
  };
  if (w.x) add(w.x);
  if (w.y) add(w.y);
  if (w.z) add(w.z);
  if (w.w) add(w.w);

  union P4 { __bf16 e[4]; uint2 u; };

  const int d = dest[bs];
  if (d >= 0) {
    const float inv = 1.0f / (float)(cnt > 0 ? cnt : 1);
    float vals[4] = {v0 * inv, v1 * inv, v2 * inv, v3 * inv};
    P4 ph, pl;
#pragma unroll
    for (int j = 0; j < 4; ++j) {
      ph.e[j] = (bf16)vals[j];
      pl.e[j] = (bf16)(vals[j] - (float)ph.e[j]);
    }
    const size_t off = (size_t)(b * SS + d) * HIDDEN + h;
    *reinterpret_cast<uint2*>(xh + off) = ph.u;
    *reinterpret_cast<uint2*>(xl + off) = pl.u;
  }
  if (s >= lens[b]) {  // pad_vec = embed[words[0, S-1, W-1]]
    const int pid = words[(SS - 1) * WW + (WW - 1)];
    const float4 pv = *reinterpret_cast<const float4*>(embed + (size_t)pid * HIDDEN + h);
    float vals[4] = {pv.x, pv.y, pv.z, pv.w};
    P4 ph, pl;
#pragma unroll
    for (int j = 0; j < 4; ++j) {
      ph.e[j] = (bf16)vals[j];
      pl.e[j] = (bf16)(vals[j] - (float)ph.e[j]);
    }
    const size_t off = (size_t)bs * HIDDEN + h;
    *reinterpret_cast<uint2*>(xh + off) = ph.u;
    *reinterpret_cast<uint2*>(xl + off) = pl.u;
  }
}

// -------- Kernel 0: split + transpose weights to [N][K] bf16 hi/lo ----------
__global__ __launch_bounds__(256) void prep_w(const float* __restrict__ Wf,
                                              bf16* __restrict__ wth,
                                              bf16* __restrict__ wtl) {
  const int idx = blockIdx.x * 256 + threadIdx.x;   // 0 .. 768*512-1
  const int k = idx / EMBED;
  const int n = idx % EMBED;
  const float w = Wf[idx];                          // coalesced (n fastest)
  const bf16 hi = (bf16)w;
  const bf16 lo = (bf16)(w - (float)hi);
  wth[(size_t)n * HIDDEN + k] = hi;
  wtl[(size_t)n * HIDDEN + k] = lo;
}

// ------------- Kernel 3: TDM-staged split-bf16 WMMA GEMM --------------------
#define TM  64
#define TN  128
#define TK  32
#define LDA 40   // 80B padded LDS rows (TDM pad: 16 dwords data + 4 dwords pad)
#define LDB 40

// Build a 2-D tile D# (ISA §8.3/8.4) and issue TENSOR_LOAD_TO_LDS.
// data_size=2B, pad_enable, pad_interval=16 dwords (64B row), pad_amount=4 dwords (16B).
__device__ inline void tdm_load_2d(u32 lds_addr, u64 gaddr,
                                   u32 tile_d0, u32 tile_d1,
                                   u32 tens_d0, u32 tens_d1, u64 stride0) {
  u32x4 g0;
  i32x8 g1;
  i32x4 g2 = {0, 0, 0, 0};
  i32x4 g3 = {0, 0, 0, 0};
  i32x8 g4 = {0, 0, 0, 0, 0, 0, 0, 0};   // extra group required by 6-arg builtin
  const u64 stride1 = stride0 * (u64)tens_d1;

  g0[0] = 1u;                                                   // count=1 (valid user D#)
  g0[1] = lds_addr;                                             // LDS byte address
  g0[2] = (u32)gaddr;                                           // global_addr[31:0]
  g0[3] = (u32)((gaddr >> 32) & 0x01FFFFFFu) | (2u << 30);      // global_addr[56:32] | type=2

  g1[0] = (int)((1u << 16)        // data_size = 2B
              | (1u << 20)        // pad_enable
              | (3u << 22)        // pad_interval = 16 dwords
              | (3u << 25));      // pad_amount   = 4 dwords
  g1[1] = (int)((tens_d0 & 0xFFFFu) << 16);                               // tensor_dim0[15:0]
  g1[2] = (int)(((tens_d0 >> 16) & 0xFFFFu) | ((tens_d1 & 0xFFFFu) << 16));
  g1[3] = (int)(((tens_d1 >> 16) & 0xFFFFu) | ((tile_d0 & 0xFFFFu) << 16));
  g1[4] = (int)(tile_d1 & 0xFFFFu);                                       // tile_dim1, tile_dim2=0
  g1[5] = (int)(stride0 & 0xFFFFFFFFu);
  g1[6] = (int)(((stride0 >> 32) & 0xFFFFu) | ((stride1 & 0xFFFFu) << 16));
  g1[7] = (int)((stride1 >> 16) & 0xFFFFFFFFu);

  __builtin_amdgcn_tensor_load_to_lds(g0, g1, g2, g3, g4, 0);
}

__global__ __launch_bounds__(256) void gemm_bias_lrelu(const bf16* __restrict__ Xh,
                                                       const bf16* __restrict__ Xl,
                                                       const bf16* __restrict__ Wth,
                                                       const bf16* __restrict__ Wtl,
                                                       const float* __restrict__ bias,
                                                       float* __restrict__ out) {
  __shared__ __align__(16) bf16 sAhi[2][TM * LDA];
  __shared__ __align__(16) bf16 sAlo[2][TM * LDA];
  __shared__ __align__(16) bf16 sBhi[2][TN * LDB];
  __shared__ __align__(16) bf16 sBlo[2][TN * LDB];

  const int tid  = threadIdx.x;
  const int lane = tid & 31;
  const int wave = tid >> 5;
  const int wm0  = (wave >> 2) * 32;
  const int wn0  = (wave & 3) * 32;
  const int mbase = blockIdx.y * TM;
  const int nbase = blockIdx.x * TN;

  v8f acc[2][2] = {};

  // One TDM descriptor per wave 0..3 per K-step (EXEC-independent, wave-uniform args).
  auto issue = [&](int step) {
    const int bsel = step & 1;
    const int kb = step * TK;
    if (wave == 0)
      tdm_load_2d((u32)(uintptr_t)&sAhi[bsel][0],
                  (u64)(uintptr_t)(Xh + (size_t)mbase * HIDDEN + kb),
                  TK, TM, HIDDEN, MM, HIDDEN);
    else if (wave == 1)
      tdm_load_2d((u32)(uintptr_t)&sAlo[bsel][0],
                  (u64)(uintptr_t)(Xl + (size_t)mbase * HIDDEN + kb),
                  TK, TM, HIDDEN, MM, HIDDEN);
    else if (wave == 2)
      tdm_load_2d((u32)(uintptr_t)&sBhi[bsel][0],
                  (u64)(uintptr_t)(Wth + (size_t)nbase * HIDDEN + kb),
                  TK, TN, HIDDEN, EMBED, HIDDEN);
    else if (wave == 3)
      tdm_load_2d((u32)(uintptr_t)&sBlo[bsel][0],
                  (u64)(uintptr_t)(Wtl + (size_t)nbase * HIDDEN + kb),
                  TK, TN, HIDDEN, EMBED, HIDDEN);
  };

  constexpr int NSTEP = HIDDEN / TK;  // 24
  issue(0);

  for (int i = 0; i < NSTEP; ++i) {
    if (i + 1 < NSTEP) issue(i + 1);          // target buffer freed by prev trailing barrier
    if (wave < 4) {
      if (i + 1 < NSTEP) __builtin_amdgcn_s_wait_tensorcnt(1);   // tile i resident, i+1 in flight
      else               __builtin_amdgcn_s_wait_tensorcnt(0);
    }
    __syncthreads();                          // publish tile i to all waves

    const int bsel = i & 1;
    const int arow = wm0 + (lane & 15);
    const int aoff = (lane >> 4) * 8;   // lane<16: K{0..7,16..23}; lane>=16: K{8..15,24..31}
    const int brow = wn0 + (lane & 15);
    const int boff = (lane >> 4) * 8;   // lane<16: K{0..7,16..23}; lane>=16: K{8..15,24..31}

    v16bf aHi[2], aLo[2], bHi[2], bLo[2];
#pragma unroll
    for (int mt = 0; mt < 2; ++mt) {
      const bf16* ph = &sAhi[bsel][(arow + mt * 16) * LDA + aoff];
      const bf16* pl = &sAlo[bsel][(arow + mt * 16) * LDA + aoff];
      const v8bf h0 = *(const v8bf*)(ph);
      const v8bf h1 = *(const v8bf*)(ph + 16);
      const v8bf l0 = *(const v8bf*)(pl);
      const v8bf l1 = *(const v8bf*)(pl + 16);
#pragma unroll
      for (int j = 0; j < 8; ++j) {
        aHi[mt][j] = h0[j]; aHi[mt][j + 8] = h1[j];
        aLo[mt][j] = l0[j]; aLo[mt][j + 8] = l1[j];
      }
    }
#pragma unroll
    for (int nt = 0; nt < 2; ++nt) {
      const bf16* ph = &sBhi[bsel][(brow + nt * 16) * LDB + boff];
      const bf16* pl = &sBlo[bsel][(brow + nt * 16) * LDB + boff];
      const v8bf h0 = *(const v8bf*)(ph);
      const v8bf h1 = *(const v8bf*)(ph + 16);
      const v8bf l0 = *(const v8bf*)(pl);
      const v8bf l1 = *(const v8bf*)(pl + 16);
#pragma unroll
      for (int j = 0; j < 8; ++j) {
        bHi[nt][j] = h0[j]; bHi[nt][j + 8] = h1[j];
        bLo[nt][j] = l0[j]; bLo[nt][j + 8] = l1[j];
      }
    }

#pragma unroll
    for (int mt = 0; mt < 2; ++mt)
#pragma unroll
      for (int nt = 0; nt < 2; ++nt) {
        acc[mt][nt] = __builtin_amdgcn_wmma_f32_16x16x32_bf16(
            false, aHi[mt], false, bHi[nt], (short)0, acc[mt][nt], false, false);
        acc[mt][nt] = __builtin_amdgcn_wmma_f32_16x16x32_bf16(
            false, aHi[mt], false, bLo[nt], (short)0, acc[mt][nt], false, false);
        acc[mt][nt] = __builtin_amdgcn_wmma_f32_16x16x32_bf16(
            false, aLo[mt], false, bHi[nt], (short)0, acc[mt][nt], false, false);
      }
    __syncthreads();                          // tile i consumed -> buffer free for i+2
  }

  // epilogue: bias + LeakyReLU(0.1); C layout: VGPR r -> M=r+8*(lane>=16), N=lane&15
#pragma unroll
  for (int mt = 0; mt < 2; ++mt)
#pragma unroll
    for (int nt = 0; nt < 2; ++nt) {
      const int n = nbase + wn0 + nt * 16 + (lane & 15);
      const float bv = bias[n];
#pragma unroll
      for (int r = 0; r < 8; ++r) {
        const int m = mbase + wm0 + mt * 16 + r + (lane >> 4) * 8;
        float v = acc[mt][nt][r] + bv;
        v = (v > 0.0f) ? v : v * 0.1f;
        out[(size_t)m * EMBED + n] = v;
      }
    }
}

// ---------------------------------------------------------------------------
extern "C" void kernel_launch(void* const* d_in, const int* in_sizes, int n_in,
                              void* d_out, int out_size, void* d_ws, size_t ws_size,
                              hipStream_t stream) {
  (void)in_sizes; (void)n_in; (void)out_size; (void)ws_size;
  const int*   words = (const int*)d_in[0];
  const float* embed = (const float*)d_in[1];
  const float* wffn  = (const float*)d_in[2];
  const float* bffn  = (const float*)d_in[3];
  float* out = (float*)d_out;

  // workspace: xh | xl (bf16 [8192][768]) | wth | wtl (bf16 [512][768]) | dest | lens
  char* ws = (char*)d_ws;
  bf16* xh  = (bf16*)(ws);
  bf16* xl  = (bf16*)(ws + (size_t)MM * HIDDEN * 2);
  bf16* wth = (bf16*)(ws + (size_t)MM * HIDDEN * 4);
  bf16* wtl = (bf16*)(ws + (size_t)MM * HIDDEN * 4 + (size_t)EMBED * HIDDEN * 2);
  int*  dest = (int*)(ws + (size_t)MM * HIDDEN * 4 + (size_t)EMBED * HIDDEN * 4);
  int*  lens = dest + MM;

  scan_valid<<<BB, SS, 0, stream>>>(words, dest, lens);
  prep_w<<<(HIDDEN * EMBED) / 256, 256, 0, stream>>>(wffn, wth, wtl);
  pool_scatter<<<MM, 192, 0, stream>>>(words, embed, dest, lens, xh, xl);
  dim3 grid(EMBED / TN, MM / TM);  // (4, 128)
  gemm_bias_lrelu<<<grid, 256, 0, stream>>>(xh, xl, wth, wtl, bffn, out);
}
